// ContrastiveVideoAudioSimilarity_53154515255725
// MI455X (gfx1250) — matrix operand order repaired and checked by reading, not compiled, not run on hardware
//
#include <hip/hip_runtime.h>
#include <hip/hip_bf16.h>
#include <math.h>

// ---------------- problem constants ----------------
#define BATCH 8
#define TT    32
#define S     256           // H*W
#define D     1024          // DV
#define NH    16
#define DH    64
#define DA    512
#define LA    2048
#define NBT   (BATCH*TT)    // 256 (b,t) tiles
#define EPSLN 1e-6f

typedef __attribute__((ext_vector_type(16))) __bf16 v16bf;
typedef __attribute__((ext_vector_type(4)))  __bf16 v4bf;
typedef __attribute__((ext_vector_type(8)))  float  v8f;

// ---------------- workspace layout (bytes) ----------------
#define WS_Q        0                         // 1024 f32            (4 KB)
#define WS_KQ       4096                      // 16*1024 bf16        (32 KB)
#define WS_POOLEDX  65536                     // 256*16*1024 f32     (16 MB)
#define WS_CTX      (WS_POOLEDX + 16777216)   // 256*1024 f32        (1 MB)
#define WS_PLN      (WS_CTX + 1048576)        // 256*1024 f32        (1 MB)
#define WS_VT       (WS_PLN + 1048576)        // 256*512 bf16        (256 KB)
#define WS_AHAT     (WS_VT + 262144)          // 8*2048*512 bf16     (16 MB)

// ================= kernel 0: q = probe @ wq^T + bq =================
__global__ void k_qvec(const float* __restrict__ probe,
                       const float* __restrict__ wq,
                       const float* __restrict__ bq,
                       float* __restrict__ qout) {
    int d = blockIdx.x * 256 + threadIdx.x;      // 0..1023
    const float4* p4 = (const float4*)probe;
    const float4* w4 = (const float4*)(wq + (size_t)d * D);
    float acc = bq[d];
    for (int e = 0; e < D / 4; ++e) {
        float4 a = p4[e], b = w4[e];
        acc += a.x * b.x + a.y * b.y + a.z * b.z + a.w * b.w;
    }
    qout[d] = acc;
}

// ============ kernel 1: kq[h,e] = sum_{d in head h} q[d]*wk[d,e]  (bf16) ============
__global__ void k_foldq(const float* __restrict__ q,
                        const float* __restrict__ wk,
                        __bf16* __restrict__ kqbf) {
    int h = blockIdx.x;                          // 0..15
    for (int rep = 0; rep < 4; ++rep) {
        int e = threadIdx.x + rep * 256;         // 0..1023
        float acc = 0.f;
        for (int d = 0; d < DH; ++d)
            acc += q[h * DH + d] * wk[(size_t)(h * DH + d) * D + e];
        kqbf[h * D + e] = (__bf16)acc;
    }
}

// ============ kernel 2: fused attention-pool per (b,t) tile ============
// pass1: scores[s,h] = X_s . kq_h (WMMA bf16)   pass2: pooledX[h,e] = attn^T @ X (WMMA bf16)
__global__ __launch_bounds__(256)
void k_attnpool(const float* __restrict__ video,
                const __bf16* __restrict__ kqbf,
                float* __restrict__ pooledX) {
    extern __shared__ char smem[];
    float*  s_scores = (float*)smem;                        // [S][NH]  16 KB
    __bf16* s_attnT  = (__bf16*)(smem + 16384);             // [NH][S]   8 KB
    __bf16* s_xchunk = (__bf16*)(smem + 16384 + 8192);      // [32][D]  64 KB

    const int bt   = blockIdx.x;
    const int tid  = threadIdx.x;
    const int wave = tid >> 5;
    const int lane = tid & 31;
    const int lo   = lane & 15;
    const int hi   = lane >> 4;              // 0 or 1
    const float* xbase = video + (size_t)bt * S * D;

    // ---------------- pass 1: scores via WMMA ----------------
    for (int t = 0; t < 2; ++t) {
        const int M0 = (wave * 2 + t) * 16;
        v8f acc = {0.f,0.f,0.f,0.f,0.f,0.f,0.f,0.f};
        const float4* xr = (const float4*)(xbase + (size_t)(M0 + lo) * D);
        for (int k0 = 0; k0 < D; k0 += 32) {
            const int kb = k0 + (hi ? 16 : 0);
            float4 p0 = xr[kb / 4], p1 = xr[kb / 4 + 1];
            float4 p2 = xr[kb / 4 + 2], p3 = xr[kb / 4 + 3];
            v16bf a;
            a[0]=(__bf16)p0.x; a[1]=(__bf16)p0.y; a[2]=(__bf16)p0.z; a[3]=(__bf16)p0.w;
            a[4]=(__bf16)p1.x; a[5]=(__bf16)p1.y; a[6]=(__bf16)p1.z; a[7]=(__bf16)p1.w;
            a[8]=(__bf16)p2.x; a[9]=(__bf16)p2.y; a[10]=(__bf16)p2.z; a[11]=(__bf16)p2.w;
            a[12]=(__bf16)p3.x; a[13]=(__bf16)p3.y; a[14]=(__bf16)p3.z; a[15]=(__bf16)p3.w;
            v16bf b = *(const v16bf*)(kqbf + lo * D + k0 + (hi ? 16 : 0));
            acc = __builtin_amdgcn_wmma_f32_16x16x32_bf16(
                false, a, false, b, (short)0, acc, false, false);
        }
        for (int j = 0; j < 8; ++j) {
            int m = M0 + j + (hi << 3);
            s_scores[m * NH + lo] = acc[j] * 0.125f;   // 1/sqrt(64)
        }
    }
    __syncthreads();

    // ---------------- softmax over S per head ----------------
    if (tid < NH) {
        const int h = tid;
        float mx = -1e30f;
        for (int s = 0; s < S; ++s) mx = fmaxf(mx, s_scores[s * NH + h]);
        float sum = 0.f;
        for (int s = 0; s < S; ++s) {
            float e = __expf(s_scores[s * NH + h] - mx);
            s_scores[s * NH + h] = e;
            sum += e;
        }
        float inv = 1.f / sum;
        for (int s = 0; s < S; ++s)
            s_attnT[h * S + s] = (__bf16)(s_scores[s * NH + h] * inv);
    }
    __syncthreads();

    // ---------------- pass 2: pooledX = attn^T @ X via WMMA ----------------
    v8f acc2[8];
    for (int nt = 0; nt < 8; ++nt)
        acc2[nt] = (v8f){0.f,0.f,0.f,0.f,0.f,0.f,0.f,0.f};

    const int lrow = tid >> 3;               // 0..31
    const int lcol = (tid & 7) * 128;        // 0..896
    for (int c = 0; c < 8; ++c) {
        const int s0 = c * 32;
        // stage 32 rows of X as bf16 into LDS (cooperative, float4 loads)
        const float4* src = (const float4*)(xbase + (size_t)(s0 + lrow) * D + lcol);
        if (c < 7) __builtin_prefetch(xbase + (size_t)(s0 + 32 + lrow) * D + lcol, 0, 0);
        for (int i = 0; i < 32; ++i) {
            float4 v = src[i];
            v4bf w; w[0]=(__bf16)v.x; w[1]=(__bf16)v.y; w[2]=(__bf16)v.z; w[3]=(__bf16)v.w;
            *(v4bf*)(s_xchunk + lrow * D + lcol + i * 4) = w;
        }
        __syncthreads();

        // A = attn^T tile [16h x 32s] for this chunk (same for all n-tiles)
        v16bf a;
        {
            const int kb = hi ? 16 : 0;
            for (int i = 0; i < 8; ++i) {
                a[i]     = s_attnT[lo * S + s0 + kb + i];
                a[8 + i] = s_attnT[lo * S + s0 + kb + 8 + i];
            }
        }
        const int ks = hi ? 16 : 0;
        for (int nt = 0; nt < 8; ++nt) {
            const int ebase = wave * 128 + nt * 16;
            v16bf b;
            for (int i = 0; i < 16; ++i)
                b[i] = s_xchunk[(ks + i) * D + ebase + lo];
            acc2[nt] = __builtin_amdgcn_wmma_f32_16x16x32_bf16(
                false, a, false, b, (short)0, acc2[nt], false, false);
        }
        __syncthreads();
    }
    // write pooledX[bt][h][e]
    for (int nt = 0; nt < 8; ++nt)
        for (int j = 0; j < 8; ++j) {
            int h = j + (hi << 3);
            int e = wave * 128 + nt * 16 + lo;
            pooledX[(size_t)bt * NH * D + h * D + e] = acc2[nt][j];
        }
}

// ============ kernel 3: ctx[bt,d] = pooledX[bt,h(d)] . wv[d,:] + bv[d] ============
__global__ void k_ctx(const float* __restrict__ pooledX,
                      const float* __restrict__ wv,
                      const float* __restrict__ bv,
                      float* __restrict__ ctx) {
    const int bt = blockIdx.x;
    for (int rep = 0; rep < 4; ++rep) {
        int d = threadIdx.x + rep * 256;
        int h = d >> 6;
        const float4* p4 = (const float4*)(pooledX + (size_t)bt * NH * D + h * D);
        const float4* w4 = (const float4*)(wv + (size_t)d * D);
        float acc = bv[d];
        for (int e = 0; e < D / 4; ++e) {
            float4 a = p4[e], b = w4[e];
            acc += a.x * b.x + a.y * b.y + a.z * b.z + a.w * b.w;
        }
        ctx[(size_t)bt * D + d] = acc;
    }
}

// ============ kernel 4: pooled = ctx@wo^T + bo, then LayerNorm ============
__global__ __launch_bounds__(256)
void k_pool_ln(const float* __restrict__ ctx,
               const float* __restrict__ wo,
               const float* __restrict__ bo,
               const float* __restrict__ lng,
               const float* __restrict__ lnb,
               float* __restrict__ pln) {
    __shared__ float s_in[D];
    __shared__ float s_out[D];
    __shared__ float red1[256], red2[256];
    const int bt = blockIdx.x, tid = threadIdx.x;
    for (int i = 0; i < 4; ++i)
        s_in[tid + i * 256] = ctx[(size_t)bt * D + tid + i * 256];
    __syncthreads();
    float psum = 0.f, psq = 0.f;
    for (int i = 0; i < 4; ++i) {
        int d = tid + i * 256;
        const float4* w4 = (const float4*)(wo + (size_t)d * D);
        const float4* c4 = (const float4*)s_in;
        float acc = bo[d];
        for (int e = 0; e < D / 4; ++e) {
            float4 a = c4[e], b = w4[e];
            acc += a.x * b.x + a.y * b.y + a.z * b.z + a.w * b.w;
        }
        s_out[d] = acc;
        psum += acc; psq += acc * acc;
    }
    red1[tid] = psum; red2[tid] = psq;
    __syncthreads();
    for (int st = 128; st > 0; st >>= 1) {
        if (tid < st) { red1[tid] += red1[tid + st]; red2[tid] += red2[tid + st]; }
        __syncthreads();
    }
    float mu = red1[0] / D;
    float var = red2[0] / D - mu * mu;
    float rinv = rsqrtf(var + EPSLN);
    for (int i = 0; i < 4; ++i) {
        int d = tid + i * 256;
        pln[(size_t)bt * D + d] = (s_out[d] - mu) * rinv * lng[d] + lnb[d];
    }
}

// ============ kernel 5: vt = LN@proj^T + pb, L2-normalize, -> bf16 ============
__global__ __launch_bounds__(256)
void k_vt(const float* __restrict__ pln,
          const float* __restrict__ pw,
          const float* __restrict__ pb,
          __bf16* __restrict__ vtbf) {
    __shared__ float s_in[D];
    __shared__ float s_out[DA];
    __shared__ float red[256];
    const int bt = blockIdx.x, tid = threadIdx.x;
    for (int i = 0; i < 4; ++i)
        s_in[tid + i * 256] = pln[(size_t)bt * D + tid + i * 256];
    __syncthreads();
    float psq = 0.f;
    for (int i = 0; i < 2; ++i) {
        int j = tid + i * 256;
        const float4* w4 = (const float4*)(pw + (size_t)j * D);
        const float4* c4 = (const float4*)s_in;
        float acc = pb[j];
        for (int e = 0; e < D / 4; ++e) {
            float4 a = c4[e], b = w4[e];
            acc += a.x * b.x + a.y * b.y + a.z * b.z + a.w * b.w;
        }
        s_out[j] = acc;
        psq += acc * acc;
    }
    red[tid] = psq;
    __syncthreads();
    for (int st = 128; st > 0; st >>= 1) {
        if (tid < st) red[tid] += red[tid + st];
        __syncthreads();
    }
    float rn = rsqrtf(red[0]);
    for (int i = 0; i < 2; ++i) {
        int j = tid + i * 256;
        vtbf[(size_t)bt * DA + j] = (__bf16)(s_out[j] * rn);
    }
}

// ============ kernel 6: audio L2-normalize -> bf16 ============
__global__ __launch_bounds__(128)
void k_ahat(const float* __restrict__ audio, __bf16* __restrict__ ahat) {
    __shared__ float red[128];
    const size_t row = blockIdx.x;           // b*LA + l  (16384 rows)
    const int tid = threadIdx.x;
    const float* src = audio + row * DA;
    float psq = 0.f;
    float v[4];
    for (int i = 0; i < 4; ++i) { v[i] = src[tid + i * 128]; psq += v[i] * v[i]; }
    red[tid] = psq;
    __syncthreads();
    for (int st = 64; st > 0; st >>= 1) {
        if (tid < st) red[tid] += red[tid + st];
        __syncthreads();
    }
    float rn = rsqrtf(red[0]);
    for (int i = 0; i < 4; ++i)
        ahat[row * DA + tid + i * 128] = (__bf16)(v[i] * rn);
}

// ============ kernel 7: sim = scale * vt @ ahat^T + bias  (WMMA bf16) ============
__global__ __launch_bounds__(32)
void k_sim(const __bf16* __restrict__ vtbf,
           const __bf16* __restrict__ ahat,
           const float* __restrict__ lscale,
           const float* __restrict__ lbias,
           float* __restrict__ out) {
    const int b  = blockIdx.z;
    const int M0 = blockIdx.y * 16;          // t-tile
    const int l0 = blockIdx.x * 16;          // l-tile
    const int lane = threadIdx.x;
    const int lo = lane & 15, hi = lane >> 4;
    v8f acc = {0.f,0.f,0.f,0.f,0.f,0.f,0.f,0.f};
    const __bf16* arow = vtbf + (size_t)(b * TT + M0 + lo) * DA;
    for (int k0 = 0; k0 < DA; k0 += 32) {
        const int kb = k0 + (hi ? 16 : 0);
        v16bf a;
        for (int i = 0; i < 8; ++i) {
            a[i]     = arow[kb + i];
            a[8 + i] = arow[kb + 8 + i];
        }
        v16bf bmat = *(const v16bf*)(ahat + ((size_t)b * LA + l0 + lo) * DA
                                     + k0 + (hi ? 16 : 0));
        acc = __builtin_amdgcn_wmma_f32_16x16x32_bf16(
            false, a, false, bmat, (short)0, acc, false, false);
    }
    const float sc = __expf(lscale[0]);
    const float bi = lbias[0];
    for (int j = 0; j < 8; ++j) {
        int t = M0 + j + (hi << 3);
        out[((size_t)b * TT + t) * LA + l0 + lo] = acc[j] * sc + bi;
    }
}

// ================================ launcher ================================
extern "C" void kernel_launch(void* const* d_in, const int* in_sizes, int n_in,
                              void* d_out, int out_size, void* d_ws, size_t ws_size,
                              hipStream_t stream) {
    const float* video = (const float*)d_in[0];
    const float* audio = (const float*)d_in[1];
    const float* probe = (const float*)d_in[2];
    const float* wq    = (const float*)d_in[3];
    const float* wk    = (const float*)d_in[4];
    const float* wv    = (const float*)d_in[5];
    const float* bq    = (const float*)d_in[6];
    // d_in[7] = bk: cancels in softmax (constant per head across S)
    const float* bv    = (const float*)d_in[8];
    const float* wo    = (const float*)d_in[9];
    const float* bo    = (const float*)d_in[10];
    const float* lng   = (const float*)d_in[11];
    const float* lnb   = (const float*)d_in[12];
    const float* pw    = (const float*)d_in[13];
    const float* pb    = (const float*)d_in[14];
    const float* lsc   = (const float*)d_in[15];
    const float* lbi   = (const float*)d_in[16];

    char* ws = (char*)d_ws;
    float*  ws_q   = (float*)(ws + WS_Q);
    __bf16* ws_kq  = (__bf16*)(ws + WS_KQ);
    float*  ws_px  = (float*)(ws + WS_POOLEDX);
    float*  ws_ctx = (float*)(ws + WS_CTX);
    float*  ws_pln = (float*)(ws + WS_PLN);
    __bf16* ws_vt  = (__bf16*)(ws + WS_VT);
    __bf16* ws_ah  = (__bf16*)(ws + WS_AHAT);
    float*  out    = (float*)d_out;

    k_qvec   <<<4,   256, 0, stream>>>(probe, wq, bq, ws_q);
    k_foldq  <<<NH,  256, 0, stream>>>(ws_q, wk, ws_kq);
    k_attnpool<<<NBT, 256, 90112, stream>>>(video, ws_kq, ws_px);
    k_ctx    <<<NBT, 256, 0, stream>>>(ws_px, wv, bv, ws_ctx);
    k_pool_ln<<<NBT, 256, 0, stream>>>(ws_ctx, wo, bo, lng, lnb, ws_pln);
    k_vt     <<<NBT, 256, 0, stream>>>(ws_pln, pw, pb, ws_vt);
    k_ahat   <<<BATCH * LA, 128, 0, stream>>>(audio, ws_ah);
    k_sim    <<<dim3(LA / 16, TT / 16, BATCH), 32, 0, stream>>>(ws_vt, ws_ah, lsc, lbi, out);
}
